// CholecAlignment_32890859552987
// MI455X (gfx1250) — 8-bit matrix operand order repaired and checked by reading, compiled, not run on hardware
//
#include <hip/hip_runtime.h>
#include <hip/hip_bf16.h>

typedef __attribute__((ext_vector_type(8))) int v8i;

#define HW_PIX   147456
#define NB       8
#define NP       32
#define NT       16
#define K_CHUNKS (HW_PIX / 64)
#define GEMM_BLOCKS_X 72
#define CHUNKS_PER_BLK (K_CHUNKS / GEMM_BLOCKS_X)

__global__ void __launch_bounds__(256) pack_kernel(
    const float* __restrict__ in, unsigned char* __restrict__ out,
    unsigned* __restrict__ area)
{
    __shared__ unsigned sred[256];
    const int mask = blockIdx.y;
    const size_t px = (size_t)mask * HW_PIX + (size_t)blockIdx.x * 1024 +
                      (size_t)threadIdx.x * 4;
    const float4 f = *(const float4*)(in + px);
    const unsigned b0 = (f.x != 0.0f) ? 1u : 0u;
    const unsigned b1 = (f.y != 0.0f) ? 1u : 0u;
    const unsigned b2 = (f.z != 0.0f) ? 1u : 0u;
    const unsigned b3 = (f.w != 0.0f) ? 1u : 0u;
    *(unsigned*)(out + px) = b0 | (b1 << 8) | (b2 << 16) | (b3 << 24);

    sred[threadIdx.x] = b0 + b1 + b2 + b3;
    __syncthreads();
    for (int s = 128; s > 0; s >>= 1) {
        if (threadIdx.x < (unsigned)s) sred[threadIdx.x] += sred[threadIdx.x + s];
        __syncthreads();
    }
    if (threadIdx.x == 0) atomicAdd(&area[mask], sred[0]);
}

__global__ void __launch_bounds__(64) wmma_inter_kernel(
    const unsigned char* __restrict__ gp, const unsigned char* __restrict__ gt,
    int* __restrict__ inter)
{
    const int n    = blockIdx.y;
    const int mt   = threadIdx.x >> 5;
    const int lane = threadIdx.x & 31;
    const int lh   = lane >> 4;
    const int m    = lane & 15;

    const unsigned* gp32 =
        (const unsigned*)(gp + (size_t)(n * NP + mt * 16 + m) * HW_PIX);
    const unsigned* gt32 =
        (const unsigned*)(gt + (size_t)(n * NT + m) * HW_PIX);

    int offA[8], offB[8];
#pragma unroll
    for (int v = 0; v < 8; ++v) {
        offA[v] = (v & 3) + 4 * lh + 8 * (v >> 2);
        offB[v] = (v & 3) + 4 * lh + 8 * (v >> 2);
    }

    const int c0 = blockIdx.x * CHUNKS_PER_BLK;
    v8i acc = {};
    for (int c = 0; c < CHUNKS_PER_BLK; ++c) {
        const int base_dw = (c0 + c) << 4;
        v8i a, b;
#pragma unroll
        for (int v = 0; v < 8; ++v) a[v] = (int)gp32[base_dw + offA[v]];
#pragma unroll
        for (int v = 0; v < 8; ++v) b[v] = (int)gt32[base_dw + offB[v]];
        acc = __builtin_amdgcn_wmma_i32_16x16x64_iu8(false, a, false, b, acc,
                                                     false, false);
    }

#pragma unroll
    for (int j = 0; j < 8; ++j) {
        const int row = mt * 16 + j + 8 * lh;
        atomicAdd(&inter[(n * NP + row) * NT + m], acc[j]);
    }
}

__global__ void __launch_bounds__(256) ioumax_kernel(
    const int* __restrict__ inter, const unsigned* __restrict__ area_p,
    const unsigned* __restrict__ area_t, float* __restrict__ ioumax)
{
    const int id = threadIdx.x;
    const int n  = id >> 5;
    const float ap = (float)area_p[id];
    float best = 0.0f;
#pragma unroll
    for (int t = 0; t < NT; ++t) {
        const float it = (float)inter[id * NT + t];
        const float un = ap + (float)area_t[n * NT + t] - it;
        const float iou = (un > 0.0f) ? (it / un) : 0.0f;
        best = fmaxf(best, iou);
    }
    ioumax[id] = best;
}

__global__ void __launch_bounds__(256) score_kernel(
    const unsigned char* __restrict__ gp, const float* __restrict__ ioumax,
    float* __restrict__ out)
{
    __shared__ float w[NP];
    const int n = blockIdx.y;
    if (threadIdx.x < NP) w[threadIdx.x] = ioumax[n * NP + threadIdx.x];
    __syncthreads();

    const int pix = blockIdx.x * 1024 + threadIdx.x * 4;
    const unsigned* base = (const unsigned*)(gp + (size_t)n * NP * HW_PIX);

    float num[4] = {0.f, 0.f, 0.f, 0.f};
    float den[4] = {0.f, 0.f, 0.f, 0.f};
#pragma unroll 4
    for (int p = 0; p < NP; ++p) {
        const unsigned word = base[(p * HW_PIX + pix) >> 2];
        const float wp = w[p];
#pragma unroll
        for (int b = 0; b < 4; ++b) {
            const float g = (float)((word >> (8 * b)) & 1u);
            num[b] += wp * g;
            den[b] += g;
        }
    }
    float4 o;
    o.x = (den[0] > 0.0f) ? (num[0] / den[0]) : 0.0f;
    o.y = (den[1] > 0.0f) ? (num[1] / den[1]) : 0.0f;
    o.z = (den[2] > 0.0f) ? (num[2] / den[2]) : 0.0f;
    o.w = (den[3] > 0.0f) ? (num[3] / den[3]) : 0.0f;
    *(float4*)(out + (size_t)n * HW_PIX + pix) = o;
}

extern "C" void kernel_launch(void* const* d_in, const int* in_sizes, int n_in,
                              void* d_out, int out_size, void* d_ws, size_t ws_size,
                              hipStream_t stream)
{
    (void)in_sizes; (void)n_in; (void)out_size; (void)ws_size;
    const float* groups_pred = (const float*)d_in[0];
    const float* groups_true = (const float*)d_in[1];
    float* out = (float*)d_out;

    const size_t szGp = (size_t)NB * NP * HW_PIX;
    const size_t szGt = (size_t)NB * NT * HW_PIX;
    unsigned char* gp_u8 = (unsigned char*)d_ws;
    unsigned char* gt_u8 = gp_u8 + szGp;
    unsigned char* acc   = gt_u8 + szGt;
    int*      inter  = (int*)acc;
    unsigned* area_p = (unsigned*)(acc + (size_t)NB * NP * NT * 4);
    unsigned* area_t = area_p + NB * NP;
    float*    ioumax = (float*)(area_t + NB * NT);

    const size_t accBytes = (size_t)NB * NP * NT * 4 + NB * NP * 4 +
                            NB * NT * 4 + NB * NP * 4;
    hipMemsetAsync(acc, 0, accBytes, stream);

    pack_kernel<<<dim3(HW_PIX / 1024, NB * NP), 256, 0, stream>>>(
        groups_pred, gp_u8, area_p);
    pack_kernel<<<dim3(HW_PIX / 1024, NB * NT), 256, 0, stream>>>(
        groups_true, gt_u8, area_t);

    wmma_inter_kernel<<<dim3(GEMM_BLOCKS_X, NB), 64, 0, stream>>>(
        gp_u8, gt_u8, inter);

    ioumax_kernel<<<1, 256, 0, stream>>>(inter, area_p, area_t, ioumax);

    score_kernel<<<dim3(HW_PIX / 1024, NB), 256, 0, stream>>>(
        gp_u8, ioumax, out);
}
